// LHNN_6356551598647
// MI455X (gfx1250) — matrix operand order repaired and checked by reading, not compiled, not run on hardware
//
#include <hip/hip_runtime.h>
#include <stdint.h>

// ---------------------------------------------------------------------------
// LHNN forward pass for MI455X (gfx1250, wave32, WMMA).
// All dense linears run through one WMMA f16-in/f32-acc GEMM kernel
// (v_wmma_f32_16x16x32_f16). K-depth is a template parameter so the A-tile
// register fragments are statically indexed (no v_movrels indirect VGPR
// addressing). Pin scatters use f32 atomics; lattice / sampled-graph segment
// sums exploit contiguous dst (repeat(arange,f)).
// ---------------------------------------------------------------------------

#define NC_ 131072
#define NN_ 32768
#define NPINS_ 524288
#define HDIM 64

typedef __attribute__((ext_vector_type(16))) _Float16 v16h;
typedef __attribute__((ext_vector_type(8)))  float    v8f;

// ---------------------------------------------------------------------------
// Generic fused GEMM:  Y[M,dout] = f( Res, concat(X1,X2) @ W + b )
//   RESMODE 0: Y = relu(XW+b)
//   RESMODE 1: Y = relu(Res + XW + b)          (HeteroRes / ResidualBlock)
//   RESMODE 2: Y = Res + relu(XW + b)          (vn/vc update, lattice head)
// X1 supplies K in [0,din1), X2 supplies K in [din1,din)   (concat-linear).
// DINP = padded K depth (32/64/128), compile-time so kt loops fully unroll.
// M must be a multiple of 64 (NC, NN both are).
// One block = 4 waves = 64 rows; W^T cached in LDS as f16, zero padded.
// ---------------------------------------------------------------------------
template <int DINP, int RESMODE>
__global__ __launch_bounds__(128)
void lhnn_gemm(const float* __restrict__ X1, const float* __restrict__ X2,
               int din1, int din,
               const float* __restrict__ W, const float* __restrict__ Bb,
               const float* __restrict__ Res,
               float* __restrict__ Y, int M, int dout)
{
    constexpr int NK = DINP / 32;          // K steps of 32 (1, 2 or 4)
    __shared__ _Float16 ldsW[128 * DINP];  // [n][k] transposed, f16, padded
    __shared__ float    ldsB[128];

    const int doutP = (dout + 15) & ~15;
    const int tid   = threadIdx.x;

    // cooperative load of W^T (f16) + bias into LDS, zero padded
    for (int idx = tid; idx < doutP * DINP; idx += 128) {
        int n = idx >> (NK == 1 ? 5 : (NK == 2 ? 6 : 7));  // idx / DINP
        int k = idx & (DINP - 1);
        float v = (k < din && n < dout) ? W[(size_t)k * dout + n] : 0.0f;
        ldsW[idx] = (_Float16)v;
    }
    for (int n = tid; n < doutP; n += 128)
        ldsB[n] = (n < dout) ? Bb[n] : 0.0f;
    __syncthreads();

    const int wave = tid >> 5;
    const int lane = tid & 31;
    const int half = lane >> 4;      // lane group (0: lanes 0-15, 1: 16-31)
    const int l16  = lane & 15;
    const int rowBase = blockIdx.x * 64 + wave * 16;
    if (rowBase >= M) return;        // M % 64 == 0 in this model
    const int aRow = rowBase + l16;  // A-matrix row for this lane
    const int din2 = din - din1;
    const int nN   = doutP >> 4;     // N tiles of 16 (<= 8)

    // --- preload this wave's 16 x DINP A-tile into registers (ISA A layout:
    //     VGPR j holds K pair k0 = (j&4)<<2 | half*8 | (j&3)*2, +kt*32).
    //     kt is compile-time -> aReg stays in named VGPRs (no movrels). -----
    v16h aReg[NK];
    #pragma unroll
    for (int kt = 0; kt < NK; ++kt) {
        v16h a;
        #pragma unroll
        for (int j = 0; j < 8; ++j) {
            int k0 = kt * 32 + ((j & 4) << 2) + half * 8 + ((j & 3) << 1);
            float x0 = 0.0f, x1 = 0.0f;
            if (k0 < din1) {
                const float* p = X1 + (size_t)aRow * din1 + k0;
                x0 = p[0]; x1 = p[1];
            } else if (k0 < din) {
                const float* p = X2 + (size_t)aRow * din2 + (k0 - din1);
                x0 = p[0]; x1 = p[1];
            }
            a[2 * j]     = (_Float16)x0;
            a[2 * j + 1] = (_Float16)x1;
        }
        aReg[kt] = a;
    }

    // --- N-tile loop: B from LDS (same K order as A:
    //     VGPR j holds K pair (j&4)<<2 | half*8 | (j&3)*2, +kt*32), accumulate via WMMA ----
    for (int nt = 0; nt < nN; ++nt) {
        const int n = nt * 16 + l16;
        v8f acc = {0.f, 0.f, 0.f, 0.f, 0.f, 0.f, 0.f, 0.f};
        #pragma unroll
        for (int kt = 0; kt < NK; ++kt) {
            union { v16h h; unsigned int u[8]; } bb;
            const unsigned int* w32 = reinterpret_cast<const unsigned int*>(
                ldsW + (size_t)n * DINP + kt * 32);
            #pragma unroll
            for (int j = 0; j < 8; ++j)
                bb.u[j] = w32[half * 4 + j + (j & 4)];      // 32-bit LDS reads (aligned)
            acc = __builtin_amdgcn_wmma_f32_16x16x32_f16(
                false, aReg[kt], false, bb.h, (short)0, acc, false, false);
        }
        // epilogue: lane holds C/D column n, rows r + 8*half (r = 0..7)
        if (n < dout) {
            const float bias = ldsB[n];
            #pragma unroll
            for (int r = 0; r < 8; ++r) {
                const int row = rowBase + r + half * 8;
                float v = acc[r] + bias;
                const size_t off = (size_t)row * dout + n;
                if (RESMODE == 1)      { v += Res[off]; v = v > 0.f ? v : 0.f; }
                else if (RESMODE == 2) { v = (v > 0.f ? v : 0.f) + Res[off]; }
                else                   { v = v > 0.f ? v : 0.f; }
                Y[off] = v;
            }
        }
    }
}

// --- random-destination segment sum (pin incidence): atomic f32 adds --------
__global__ __launch_bounds__(256)
void lhnn_scatter_add(const float* __restrict__ X, const int* __restrict__ srcIdx,
                      const int* __restrict__ dstIdx, float* __restrict__ Y,
                      int npins)
{
    int t = blockIdx.x * blockDim.x + threadIdx.x;
    if (t >= npins * HDIM) return;
    int p = t >> 6;          // pin
    int f = t & 63;          // feature
    atomicAdd(&Y[(size_t)dstIdx[p] * HDIM + f], X[(size_t)srcIdx[p] * HDIM + f]);
}

// --- fixed-fanout segment sum (dst = repeat(arange(N), fan)): gather, no
//     atomics. Y[i] = (Base ? Base[i] : 0) + sum_j X[src[i*fan+j]] -----------
__global__ __launch_bounds__(256)
void lhnn_gather_sum(const float* __restrict__ X, const int* __restrict__ src,
                     const float* __restrict__ Base, float* __restrict__ Y,
                     int N, int F, int fan)
{
    int t = blockIdx.x * blockDim.x + threadIdx.x;
    if (t >= N * F) return;
    int node = t / F;
    int f    = t - node * F;
    float acc = Base ? Base[t] : 0.0f;
    const int* s = src + (size_t)node * fan;
    for (int j = 0; j < fan; ++j)
        acc += X[(size_t)s[j] * F + f];
    Y[t] = acc;
}

// --- final sigmoid + transpose: out[k] = sigmoid(V[k % NC][k / NC]) ---------
__global__ __launch_bounds__(256)
void lhnn_sigmoid_t(const float* __restrict__ V, float* __restrict__ out,
                    int ncells, int nout)
{
    int k = blockIdx.x * blockDim.x + threadIdx.x;
    if (k >= ncells * nout) return;
    int node = k % ncells;
    int ch   = k / ncells;
    float v = V[(size_t)node * nout + ch];
    out[k] = 1.0f / (1.0f + __expf(-v));
}

// ---------------------------------------------------------------------------
// Host-side orchestration
// ---------------------------------------------------------------------------
static void gemm(hipStream_t st, const float* X1, const float* X2,
                 int din1, int din, const float* W, const float* B,
                 const float* Res, int resMode, float* Y, int M, int dout)
{
    dim3 g(M / 64), b(128);
    #define LGEMM(DP, RM) \
        lhnn_gemm<DP, RM><<<g, b, 0, st>>>(X1, X2, din1, din, W, B, Res, Y, M, dout)
    if (din <= 32) {
        if (resMode == 0) LGEMM(32, 0); else if (resMode == 1) LGEMM(32, 1); else LGEMM(32, 2);
    } else if (din <= 64) {
        if (resMode == 0) LGEMM(64, 0); else if (resMode == 1) LGEMM(64, 1); else LGEMM(64, 2);
    } else {
        if (resMode == 0) LGEMM(128, 0); else if (resMode == 1) LGEMM(128, 1); else LGEMM(128, 2);
    }
    #undef LGEMM
}

static inline void gather(hipStream_t st, const float* X, const int* src,
                          const float* Base, float* Y, int N, int F, int fan)
{
    int total = N * F;
    lhnn_gather_sum<<<(total + 255) / 256, 256, 0, st>>>(X, src, Base, Y, N, F, fan);
}

static const size_t NCF = (size_t)NC_ * HDIM;  // cell feature slot (floats)
static const size_t NNF = (size_t)NN_ * HDIM;  // net  feature slot (floats)

// One HyperMP block. P = param leaf pointers, pb = base leaf index.
// hc/hn are in/out slots; h1c/h1n read-only; s*/m* scratch; cw/nw wide scratch.
static void run_hyper(hipStream_t st, const float* const* P, int pb,
                      float* hc, float* hn, const float* h1c, const float* h1n,
                      float* s1, float* s2, float* s3, float* s4,
                      float* m1, float* m2, float* m3, float* cw, float* nw,
                      const int* pin_cell, const int* pin_net,
                      const int* es, int fan)
{
    // res1 (HeteroResidualBlock)
    gemm(st, hc, nullptr, 64, 64,  P[pb+12], P[pb+13], nullptr, 0, cw, NC_, 128);
    gemm(st, cw, nullptr, 128,128, P[pb+16], P[pb+17], hc,      1, s1, NC_, 64);   // hc' = s1
    gemm(st, hn, nullptr, 64, 64,  P[pb+14], P[pb+15], nullptr, 0, nw, NN_, 128);
    gemm(st, nw, nullptr, 128,128, P[pb+18], P[pb+19], hn,      1, m1, NN_, 64);   // hn' = m1
    // vn1 = relu(h1n @ n1),  vc1 = relu(h1c @ c1)
    gemm(st, h1n, nullptr, 64,64, P[pb+6], P[pb+7], nullptr, 0, m2, NN_, 64);
    gemm(st, h1c, nullptr, 64,64, P[pb+0], P[pb+1], nullptr, 0, s2, NC_, 64);
    // z = segment_sum(hc'[pin_cell] -> pin_net)
    hipMemsetAsync(m3, 0, NNF * sizeof(float), st);
    lhnn_scatter_add<<<(NPINS_ * HDIM + 255) / 256, 256, 0, st>>>(s1, pin_cell, pin_net, m3, NPINS_);
    // t = relu(z @ cn1)   (reuse nw as a 64-wide net buffer)
    gemm(st, m3, nullptr, 64,64, P[pb+2], P[pb+3], nullptr, 0, nw, NN_, 64);
    // vn = hn' + relu(concat(vn1, t) @ cn2 + b)
    gemm(st, m2, nw, 64, 128, P[pb+4], P[pb+5], m1, 2, m3, NN_, 64);
    // vn = ResidualBlock(vn, res_n)  -> hn (slot reuse)
    gemm(st, m3, nullptr, 64,64,   P[pb+24], P[pb+25], nullptr, 0, nw, NN_, 128);
    gemm(st, nw, nullptr, 128,128, P[pb+26], P[pb+27], m3,      1, hn, NN_, 64);
    // vc = ResidualBlock(hc', res_c)
    gemm(st, s1, nullptr, 64,64,   P[pb+20], P[pb+21], nullptr, 0, cw, NC_, 128);
    gemm(st, cw, nullptr, 128,128, P[pb+22], P[pb+23], s1,      1, s3, NC_, 64);
    // z2 = segment_sum(vn[pin_net] -> pin_cell)
    hipMemsetAsync(s4, 0, NCF * sizeof(float), st);
    lhnn_scatter_add<<<(NPINS_ * HDIM + 255) / 256, 256, 0, st>>>(hn, pin_net, pin_cell, s4, NPINS_);
    // t2 = relu(z2 @ nc1)   (reuse cw as a 64-wide cell buffer)
    gemm(st, s4, nullptr, 64,64, P[pb+8], P[pb+9], nullptr, 0, cw, NC_, 64);
    // vc = vc + relu(concat(vc1, t2) @ nc2 + b)
    gemm(st, s2, cw, 64, 128, P[pb+10], P[pb+11], s3, 2, s4, NC_, 64);
    // hc_out = vc + segment_sum(vc[es] -> contiguous dst)
    gather(st, s4, es, s4, hc, NC_, 64, fan);
}

extern "C" void kernel_launch(void* const* d_in, const int* in_sizes, int n_in,
                              void* d_out, int out_size, void* d_ws, size_t ws_size,
                              hipStream_t stream)
{
    (void)in_sizes; (void)out_size; (void)ws_size;
    if (n_in < 94) return;   // 14 data inputs + 80 param leaves expected

    const float* h_c = (const float*)d_in[0];   // [NC,16]
    const float* h_n = (const float*)d_in[1];   // [NN,16]
    const int* pin_cell = (const int*)d_in[2];
    const int* pin_net  = (const int*)d_in[3];
    const int* la_src   = (const int*)d_in[4];
    const int* e1_src   = (const int*)d_in[6];
    const int* e2_src   = (const int*)d_in[8];
    const int* e3_src   = (const int*)d_in[10];
    const int* e4_src   = (const int*)d_in[12];

    // param leaves, jax pytree order (dict keys sorted; (W,b) tuples in order)
    const float* P[80];
    for (int i = 0; i < 80; ++i) P[i] = (const float*)d_in[14 + i];
    // 0-7:  emb1_c, emb1_n, emb2_c, emb2_n
    // 8-15: fg_res (l1c, l1n, l2c, l2n)
    // 16-43: hyp1  (c1,cn1,cn2,n1,nc1,nc2, res1{l1c,l1n,l2c,l2n}, res_c{l1,l2}, res_n{l1,l2})
    // 44-71: hyp2  (same layout)
    // 72-79: lat   (l1, l2, res1{l1,l2})

    // workspace carve: 6 cell64 slots, 1 cell128, 5 net64 slots, 1 net128
    float* base = (float*)d_ws;
    float* c0 = base;             float* c1 = c0 + NCF;  float* c2 = c1 + NCF;
    float* c3 = c2 + NCF;         float* c4 = c3 + NCF;  float* c5 = c4 + NCF;
    float* cw = c5 + NCF;                                  // NC x 128
    float* n0 = cw + 2 * NCF;     float* n1 = n0 + NNF;  float* n2 = n1 + NNF;
    float* n3 = n2 + NNF;         float* n4 = n3 + NNF;
    float* nw = n4 + NNF;                                  // NN x 128

    // ---------------- FeatureGen ----------------
    gemm(stream, h_c, nullptr, 16, 16, P[0], P[1], nullptr, 0, c0, NC_, 64);
    gemm(stream, h_n, nullptr, 16, 16, P[2], P[3], nullptr, 0, n0, NN_, 64);
    // fg_res hetero residual
    gemm(stream, c0, nullptr, 64, 64,  P[8],  P[9],  nullptr, 0, cw, NC_, 128);
    gemm(stream, cw, nullptr, 128,128, P[12], P[13], c0,      1, c2, NC_, 64);
    gemm(stream, n0, nullptr, 64, 64,  P[10], P[11], nullptr, 0, nw, NN_, 128);
    gemm(stream, nw, nullptr, 128,128, P[14], P[15], n0,      1, n2, NN_, 64);
    // per-cell sum over incident nets: c3 = segment_sum(hn[pin_net] -> pin_cell)
    hipMemsetAsync(c3, 0, NCF * sizeof(float), stream);
    lhnn_scatter_add<<<(NPINS_ * HDIM + 255) / 256, 256, 0, stream>>>(n2, pin_net, pin_cell, c3, NPINS_);
    // hc = relu(concat(hc, c3) @ emb2_c + b),  hn = relu(hn @ emb2_n + b)
    gemm(stream, c2, c3, 64, 128, P[4], P[5], nullptr, 0, c4, NC_, 64);
    gemm(stream, n2, nullptr, 64, 64, P[6], P[7], nullptr, 0, n0, NN_, 64);
    // hc = hc + segment_sum(hc[e1_src]) (fanout 6, contiguous dst)
    gather(stream, c4, e1_src, c4, c0, NC_, 64, 6);
    // snapshot h1c / h1n
    hipMemcpyAsync(c1, c0, NCF * sizeof(float), hipMemcpyDeviceToDevice, stream);
    hipMemcpyAsync(n1, n0, NNF * sizeof(float), hipMemcpyDeviceToDevice, stream);

    // ---------------- HyperMP x2 ----------------
    run_hyper(stream, P, 16, c0, n0, c1, n1, c2, c3, c4, c5, n2, n3, n4, cw, nw,
              pin_cell, pin_net, e2_src, 3);
    run_hyper(stream, P, 44, c0, n0, c1, n1, c2, c3, c4, c5, n2, n3, n4, cw, nw,
              pin_cell, pin_net, e3_src, 3);

    // ---------------- LatticeMP ----------------
    // vc = ResidualBlock(hc, lat.res1)
    gemm(stream, c0, nullptr, 64, 64,  P[76], P[77], nullptr, 0, cw, NC_, 128);
    gemm(stream, cw, nullptr, 128,128, P[78], P[79], c0,      1, c2, NC_, 64);
    // lattice neighbourhood sum (fanout 4, contiguous dst, no base)
    gather(stream, c2, la_src, nullptr, c3, NC_, 64, 4);
    // vc = relu(vc @ l1 + b) + relu(sum @ l2 + b)
    float* d0 = c4;
    float* d1 = c4 + (size_t)NC_ * 4;
    float* d2 = c4 + (size_t)NC_ * 8;
    gemm(stream, c2, nullptr, 64, 64, P[72], P[73], nullptr, 0, d0, NC_, 4);
    gemm(stream, c3, nullptr, 64, 64, P[74], P[75], d0,      2, d1, NC_, 4);
    // vc = vc + segment_sum(vc[e4_src]) (fanout 2, F=4)
    gather(stream, d1, e4_src, d1, d2, NC_, 4, 2);
    // sigmoid + transpose to [B, OUT, IMG, IMG]
    lhnn_sigmoid_t<<<(NC_ * 4 + 255) / 256, 256, 0, stream>>>(d2, (float*)d_out, NC_, 4);
}
